// Dense_25202868092933
// MI455X (gfx1250) — matrix operand order repaired and checked by reading, not compiled, not run on hardware
//
#include <hip/hip_runtime.h>
#include <hip/hip_bf16.h>

typedef __attribute__((ext_vector_type(16))) __bf16 v16bf;
typedef __attribute__((ext_vector_type(8)))  float  v8f;

#define B_SZ   64
#define F_IN   4096
#define F_OUT  4096
#define T_SZ   64
#define MT     128          // output rows per block
#define KT     32           // K step per iteration
#define LDW    40           // padded row length (bf16 elems): 80B, 16B aligned, stride 20 dwords
#define THRESHOLD (-0.5f)

union AFrag { v16bf v; uint4 q[2]; };

__global__ __launch_bounds__(256) void dense_snn_kernel(
    const float* __restrict__ X,   // [B, F_IN, T]
    const float* __restrict__ W,   // [F_OUT, F_IN]
    float* __restrict__ Out)       // [B, F_OUT, T]
{
    __shared__ alignas(16) __bf16 sW[MT][LDW];      // W tile, row-major [m][k]
    __shared__ alignas(16) __bf16 sXT[T_SZ][LDW];   // X tile transposed [t][k]
    __shared__ float sC[MT][T_SZ + 1];              // f32 GEMM result tile
    __shared__ float sSum[MT];                      // per-row sum of W

    const int tid  = threadIdx.x;
    const int lane = tid & 31;
    const int wave = tid >> 5;
    const int mblk = blockIdx.x;         // 0..31
    const int b    = blockIdx.y;         // 0..63
    const int m_base = mblk * MT;

    if (tid < MT) sSum[tid] = 0.0f;

    // ---- per-thread tile-load assignments (constant across K loop) ----
    // W tile: 2 threads per row, 16 contiguous floats each
    const int wrow = tid >> 1;
    const int wkh  = (tid & 1) * 16;
    // X tile: 8 threads per k-line, 8 contiguous t each
    const int xk   = tid >> 3;
    const int xtp  = (tid & 7) * 8;

    float wsum = 0.0f;

    // ---- WMMA fragment source offsets (contiguous K order k = 16h + s, wave32) ----
    const int arow = wave * 16 + (lane & 15);
    const int ak0  = (lane < 16) ? 0 : 16;    // A: K 0-15 vs 16-31
    const int bk0  = (lane < 16) ? 0 : 16;    // B: K 0-15 vs 16-31
    const int bcol = lane & 15;

    v8f acc0 = {}, acc1 = {}, acc2 = {}, acc3 = {};

    for (int kk = 0; kk < F_IN / KT; ++kk) {
        __syncthreads();   // previous iteration's WMMA reads done

        // ---- stage W tile (f32 -> bf16), accumulate exact f32 row sum ----
        {
            const float* g = W + (size_t)(m_base + wrow) * F_IN + kk * KT + wkh;
            float4 f0 = ((const float4*)g)[0];
            float4 f1 = ((const float4*)g)[1];
            float4 f2 = ((const float4*)g)[2];
            float4 f3 = ((const float4*)g)[3];
            wsum += f0.x + f0.y + f0.z + f0.w + f1.x + f1.y + f1.z + f1.w
                  + f2.x + f2.y + f2.z + f2.w + f3.x + f3.y + f3.z + f3.w;
            __bf16 h[16];
            h[0]=(__bf16)f0.x; h[1]=(__bf16)f0.y; h[2]=(__bf16)f0.z; h[3]=(__bf16)f0.w;
            h[4]=(__bf16)f1.x; h[5]=(__bf16)f1.y; h[6]=(__bf16)f1.z; h[7]=(__bf16)f1.w;
            h[8]=(__bf16)f2.x; h[9]=(__bf16)f2.y; h[10]=(__bf16)f2.z; h[11]=(__bf16)f2.w;
            h[12]=(__bf16)f3.x; h[13]=(__bf16)f3.y; h[14]=(__bf16)f3.z; h[15]=(__bf16)f3.w;
            *(uint4*)&sW[wrow][wkh]     = *(uint4*)&h[0];
            *(uint4*)&sW[wrow][wkh + 8] = *(uint4*)&h[8];
        }

        // ---- stage X tile transposed: sXT[t][k] = X[b, kk*32+k, t] ----
        {
            const float* g = X + ((size_t)b * F_IN + kk * KT + xk) * T_SZ + xtp;
            float4 f0 = ((const float4*)g)[0];
            float4 f1 = ((const float4*)g)[1];
            sXT[xtp + 0][xk] = (__bf16)f0.x;
            sXT[xtp + 1][xk] = (__bf16)f0.y;
            sXT[xtp + 2][xk] = (__bf16)f0.z;
            sXT[xtp + 3][xk] = (__bf16)f0.w;
            sXT[xtp + 4][xk] = (__bf16)f1.x;
            sXT[xtp + 5][xk] = (__bf16)f1.y;
            sXT[xtp + 6][xk] = (__bf16)f1.z;
            sXT[xtp + 7][xk] = (__bf16)f1.w;
        }

        __syncthreads();

        // ---- WMMA: each wave does a 16x64 strip, 4 tiles of N=16 ----
        AFrag a;
        a.q[0] = *(const uint4*)&sW[arow][ak0];        // K run 1 (8 elems, 16B)
        a.q[1] = *(const uint4*)&sW[arow][ak0 + 8];    // K run 2

        AFrag bf;
        bf.q[0] = *(const uint4*)&sXT[bcol +  0][bk0];
        bf.q[1] = *(const uint4*)&sXT[bcol +  0][bk0 + 8];
        acc0 = __builtin_amdgcn_wmma_f32_16x16x32_bf16(false, a.v, false, bf.v,
                                                       (short)0, acc0, false, false);
        bf.q[0] = *(const uint4*)&sXT[bcol + 16][bk0];
        bf.q[1] = *(const uint4*)&sXT[bcol + 16][bk0 + 8];
        acc1 = __builtin_amdgcn_wmma_f32_16x16x32_bf16(false, a.v, false, bf.v,
                                                       (short)0, acc1, false, false);
        bf.q[0] = *(const uint4*)&sXT[bcol + 32][bk0];
        bf.q[1] = *(const uint4*)&sXT[bcol + 32][bk0 + 8];
        acc2 = __builtin_amdgcn_wmma_f32_16x16x32_bf16(false, a.v, false, bf.v,
                                                       (short)0, acc2, false, false);
        bf.q[0] = *(const uint4*)&sXT[bcol + 48][bk0];
        bf.q[1] = *(const uint4*)&sXT[bcol + 48][bk0 + 8];
        acc3 = __builtin_amdgcn_wmma_f32_16x16x32_bf16(false, a.v, false, bf.v,
                                                       (short)0, acc3, false, false);
    }

    // ---- finish row sums ----
    atomicAdd(&sSum[wrow], wsum);

    // ---- spill accumulators: C layout lane L: n = L%16, rows r + (L<16?0:8) ----
    {
        const int rbase = wave * 16 + ((lane < 16) ? 0 : 8);
        const int n = lane & 15;
#pragma unroll
        for (int r = 0; r < 8; ++r) {
            sC[rbase + r][n +  0] = acc0[r];
            sC[rbase + r][n + 16] = acc1[r];
            sC[rbase + r][n + 32] = acc2[r];
            sC[rbase + r][n + 48] = acc3[r];
        }
    }

    __syncthreads();

    // ---- fused leaky-integrate-fire recurrence over T, one thread per row ----
    if (tid < MT) {
        const int row = tid;
        const float SCALING = (float)(5.0e8 * (1.0 / 3.14159265358979323846) * 1e-9 / 2.0);
        float nc = (1.0f + sSum[row]) / SCALING;
        nc = fmaxf(nc, 0.0f);
        const float scale = -1.0f / (1.0f + nc);
        const float mult  = (nc - 1.0f) / (nc + 1.0f);

        float mem = 0.0f;
        float* orow = Out + ((size_t)b * F_OUT + (m_base + row)) * T_SZ;
        const float* crow = &sC[row][0];
#pragma unroll 4
        for (int t4 = 0; t4 < T_SZ / 4; ++t4) {
            float v[4];
#pragma unroll
            for (int j = 0; j < 4; ++j) {
                float x = crow[t4 * 4 + j];
                mem = mult * mem + scale * x;
                float spk = (THRESHOLD - mem >= 0.0f) ? 1.0f : 0.0f;
                mem *= (1.0f - spk);
                v[j] = spk;
            }
            float4 s; s.x = v[0]; s.y = v[1]; s.z = v[2]; s.w = v[3];
            ((float4*)orow)[t4] = s;
        }
    }
}

extern "C" void kernel_launch(void* const* d_in, const int* in_sizes, int n_in,
                              void* d_out, int out_size, void* d_ws, size_t ws_size,
                              hipStream_t stream) {
    const float* X = (const float*)d_in[0];   // [64, 4096, 64] f32
    const float* W = (const float*)d_in[1];   // [4096, 4096] f32
    float* Out = (float*)d_out;               // [64, 4096, 64] f32
    (void)in_sizes; (void)n_in; (void)out_size; (void)d_ws; (void)ws_size;

    dim3 grid(F_OUT / MT, B_SZ);
    dim3 block(256);
    dense_snn_kernel<<<grid, block, 0, stream>>>(X, W, Out);
}
